// DGCNN_25726854103099
// MI455X (gfx1250) — matrix operand order repaired and checked by reading, not compiled, not run on hardware
//
#include <hip/hip_runtime.h>

// DGCNN forward for MI455X (gfx1250, wave32, WMMA), round 3.
// - All GEMMs: v_wmma_f32_16x16x32_f16, each wave computes a 16x64 tile
//   (4 accumulators share one A fragment -> 4 WMMAs per K-step per A load).
// - Edge-conv: gathered edge-feature A tile staged once per block in LDS in
//   fragment layout (block = O/64 waves, all share the same M tile).
// - max-over-k folded into the GEMM via a single branch-free
//   global_atomic_max_u32 on a monotonic uint key (no exec divergence).
// - BN stats accumulated with atomics in the GEMM; BN+leaky decode pass
//   restores floats. Division by 20 strength-reduced to a running remainder.

#define B_    16
#define N_    1024
#define KNN   20
#define LDCAT 512
#define BNPTS (B_ * N_)

typedef __attribute__((ext_vector_type(16))) _Float16 v16h;
typedef __attribute__((ext_vector_type(8)))  float    v8f;

__device__ __forceinline__ float leaky_f(float v) { return v >= 0.f ? v : 0.2f * v; }

// Monotonic order-preserving float -> uint key (single atomic, no branches).
__device__ __forceinline__ unsigned fkey(float v) {
    unsigned b = __float_as_uint(v);
    return (b & 0x80000000u) ? ~b : (b | 0x80000000u);
}
__device__ __forceinline__ float funkey(unsigned u) {
    return (u & 0x80000000u) ? __uint_as_float(u & 0x7FFFFFFFu)
                             : __uint_as_float(~u);
}

// ---------------------------------------------------------------------------
// kNN: one thread per (b, i). xi cached in registers; xj read as float4 when
// C%4==0. Fully unrolled top-20 insertion network stays in VGPRs.
// ---------------------------------------------------------------------------
template <int C>
__global__ void knn_kernel(const float* __restrict__ X, int ldx,
                           int* __restrict__ nbr) {
    int t = blockIdx.x * blockDim.x + threadIdx.x;
    if (t >= BNPTS) return;
    int b = t / N_;
    const float* xip = X + (size_t)t * ldx;
    const float* xb  = X + (size_t)b * N_ * ldx;

    float xi[C];
#pragma unroll
    for (int c = 0; c < C; ++c) xi[c] = xip[c];

    float bd[KNN]; int bi[KNN];
#pragma unroll
    for (int q = 0; q < KNN; ++q) { bd[q] = 3.4e38f; bi[q] = 0; }

    for (int j = 0; j < N_; ++j) {
        const float* xj = xb + (size_t)j * ldx;
        float d2 = 0.f;
        if constexpr (C % 4 == 0) {
#pragma unroll
            for (int c = 0; c < C; c += 4) {
                float4 v = *(const float4*)(xj + c);
                float d0 = xi[c] - v.x, d1 = xi[c + 1] - v.y;
                float d2_ = xi[c + 2] - v.z, d3 = xi[c + 3] - v.w;
                d2 += d0 * d0 + d1 * d1 + d2_ * d2_ + d3 * d3;
            }
        } else {
#pragma unroll
            for (int c = 0; c < C; ++c) { float df = xi[c] - xj[c]; d2 += df * df; }
        }
        if (d2 < bd[KNN - 1]) {
            float dc = d2; int ic = j;
#pragma unroll
            for (int q = 0; q < KNN; ++q) {
                if (dc < bd[q]) {
                    float td = bd[q]; int ti = bi[q];
                    bd[q] = dc; bi[q] = ic; dc = td; ic = ti;
                }
            }
        }
    }
#pragma unroll
    for (int q = 0; q < KNN; ++q) nbr[(size_t)t * KNN + q] = bi[q];
}

// ---------------------------------------------------------------------------
// Edge-conv GEMM. Block = O/64 waves; one block per 16-row M tile. The
// gathered edge-feature A tile (16x32 f16) is built once per K-step in LDS in
// exact per-lane fragment layout; each wave then runs 4 WMMAs (16x64 cols)
// off a single 32-byte LDS fragment read. Grid exact -> barriers safe.
// ---------------------------------------------------------------------------
template <int C, int O>
__global__ void edge_gemm_wmma(const float* __restrict__ X, int ldx,
                               const int* __restrict__ nbr,
                               const float* __restrict__ W,
                               const float* __restrict__ bias,
                               float* __restrict__ out, int ldo,
                               float* __restrict__ stats) {
    constexpr int Ktot   = 2 * C;
    constexpr int KSTEPS = (Ktot + 31) / 32;
    constexpr int WPB    = O / 64;         // waves per block (1, 2 or 4)
    constexpr int NSH    = 16 / WPB;       // halfs staged per thread per step

    __shared__ alignas(32) _Float16 As[512];   // 32 lanes x 16 halfs

    const int tid  = threadIdx.x;
    const int wid  = tid >> 5;
    const int lane = tid & 31;
    const int tm   = blockIdx.x;           // M tile
    const int grp  = lane >> 4;

    // Staging role: thread covers halfs [tid*NSH, tid*NSH+NSH) of the tile;
    // fragment lane Ls is constant for the whole run (NSH divides 16).
    const int Ls    = (tid * NSH) >> 4;
    const int hBase = (tid * NSH) & 15;
    const int grpS  = Ls >> 4;
    const int rowS  = tm * 16 + (Ls & 15);
    const int pntS  = rowS / KNN;          // b*N + n
    const int bS    = pntS / N_;
    const int jS    = nbr[rowS];
    const float* xiS = X + (size_t)pntS * ldx;
    const float* xjS = X + ((size_t)bS * N_ + jS) * ldx;

    const int colBase = wid * 64 + (lane & 15);

    v8f acc[4] = {};
#pragma unroll
    for (int ks = 0; ks < KSTEPS; ++ks) {
        const int kb = ks * 32;
        float sv[NSH];
#pragma unroll
        for (int q = 0; q < NSH / 4; ++q) {
            const int h0  = hBase + q * 4;
            const int ka0 = kb + h0 + (grpS << 4);
            if constexpr (C % 8 == 0) {
                // Branchless region select: feat = A - sel*B, pointer selects.
                const float* pA = (ka0 < C) ? (xjS + ka0) : (xiS + ka0 - C);
                const float* pB = xiS + ((ka0 < C) ? ka0 : ka0 - C);
                const float sel = (ka0 < C) ? 1.f : 0.f;
                float4 va = *(const float4*)pA;
                float4 vb = *(const float4*)pB;
                sv[q * 4 + 0] = va.x - sel * vb.x;
                sv[q * 4 + 1] = va.y - sel * vb.y;
                sv[q * 4 + 2] = va.z - sel * vb.z;
                sv[q * 4 + 3] = va.w - sel * vb.w;
            } else {
                // C=3: clamped addresses + selects (no divergent loads).
#pragma unroll
                for (int u = 0; u < 4; ++u) {
                    int ka  = ka0 + u;
                    int kaC = (ka < C) ? ka : 0;
                    int kiC = (ka >= C && ka < Ktot) ? (ka - C) : 0;
                    float dj = xjS[kaC] - xiS[kaC];
                    float xv = xiS[kiC];
                    sv[q * 4 + u] = (ka < C) ? dj : ((ka < Ktot) ? xv : 0.f);
                }
            }
        }
        __syncthreads();                   // previous tile fully consumed
#pragma unroll
        for (int e = 0; e < NSH; ++e) As[Ls * 16 + hBase + e] = (_Float16)sv[e];
        __syncthreads();                   // tile visible to all waves

        v16h af = *(const v16h*)(As + lane * 16);   // 32B LDS vector read

#pragma unroll
        for (int ct = 0; ct < 4; ++ct) {
            const int col = colBase + ct * 16;
            v16h bfrag;
            if constexpr (Ktot % 32 == 0) {
                const float* wr = W + (size_t)col * Ktot + kb + (grp << 4);
                __builtin_prefetch(wr + 32, 0, 1);   // global_prefetch_b8
                const float4* wp = (const float4*)wr;
                float4 w0 = wp[0], w1 = wp[1], w2 = wp[2], w3 = wp[3];
                bfrag[0]=(_Float16)w0.x; bfrag[1]=(_Float16)w0.y; bfrag[2]=(_Float16)w0.z; bfrag[3]=(_Float16)w0.w;
                bfrag[4]=(_Float16)w1.x; bfrag[5]=(_Float16)w1.y; bfrag[6]=(_Float16)w1.z; bfrag[7]=(_Float16)w1.w;
                bfrag[8]=(_Float16)w2.x; bfrag[9]=(_Float16)w2.y; bfrag[10]=(_Float16)w2.z; bfrag[11]=(_Float16)w2.w;
                bfrag[12]=(_Float16)w3.x; bfrag[13]=(_Float16)w3.y; bfrag[14]=(_Float16)w3.z; bfrag[15]=(_Float16)w3.w;
            } else {
                const float* wr = W + (size_t)col * Ktot;
#pragma unroll
                for (int h = 0; h < 16; ++h) {
                    int kw  = kb + h + (grp << 4);
                    int kwC = (kw < Ktot) ? kw : 0;
                    float bv = wr[kwC];
                    bfrag[h] = (_Float16)((kw < Ktot) ? bv : 0.f);
                }
            }
            acc[ct] = __builtin_amdgcn_wmma_f32_16x16x32_f16(
                false, af, false, bfrag, (short)0, acc[ct], false, false);
        }
    }

    // Epilogue: one division, then running remainder; branch-free atomic max.
    const int m0      = tm * 16 + (grp << 3);
    const int p2base  = m0 / KNN;
    const int rembase = m0 - p2base * KNN;
#pragma unroll
    for (int ct = 0; ct < 4; ++ct) {
        const int col = colBase + ct * 16;
        const float bsv = bias[col];
        float ps = 0.f, pq = 0.f;
        int p2 = p2base, rem = rembase;
#pragma unroll
        for (int r8 = 0; r8 < 8; ++r8) {
            float v = acc[ct][r8] + bsv;
            ps += v; pq += v * v;
            atomicMax((unsigned int*)&out[(size_t)p2 * ldo + col], fkey(v));
            if (++rem == KNN) { rem = 0; ++p2; }
        }
        atomicAdd(&stats[col],     ps);
        atomicAdd(&stats[O + col], pq);
    }
}

// ---------------------------------------------------------------------------
// Generic GEMM: Y = X * W^T + bias, optional BN stats. One wave per 16x64
// tile (4 accumulators share one A fragment). R always a multiple of 16.
// Padded output columns: clamped load addresses, guarded stores only.
// ---------------------------------------------------------------------------
template <int Ktot>
__global__ void gemm_wmma(const float* __restrict__ X, int ldx, int R,
                          const float* __restrict__ W,
                          const float* __restrict__ bias,
                          int O, float* __restrict__ Y, int ldy,
                          float* __restrict__ stats) {
    static_assert(Ktot % 32 == 0, "K must be padded to 32");
    const int og   = (O + 63) >> 6;                  // 64-col groups
    const int wave = blockIdx.x * (blockDim.x >> 5) + (threadIdx.x >> 5);
    if (wave >= og * (R >> 4)) return;               // wave-uniform, no barriers
    const int lane = threadIdx.x & 31;
    const int tm = wave / og, tg = wave % og;
    const int grp = lane >> 4;
    const int row = tm * 16 + (lane & 15);
    const float* xr = X + (size_t)row * ldx;
    const int colBase = tg * 64 + (lane & 15);

    v8f acc[4] = {};
    for (int kb = 0; kb < Ktot; kb += 32) {
        __builtin_prefetch(xr + kb + 32, 0, 1);
        // A fragment: one contiguous 16-float run at kb + 16*grp.
        const float4* pa0 = (const float4*)(xr + kb + (grp << 4));
        const float4* pa1 = (const float4*)(xr + kb + 8 + (grp << 4));
        float4 a0 = pa0[0], a1 = pa0[1], a2 = pa1[0], a3 = pa1[1];
        v16h af;
        af[0]=(_Float16)a0.x; af[1]=(_Float16)a0.y; af[2]=(_Float16)a0.z; af[3]=(_Float16)a0.w;
        af[4]=(_Float16)a1.x; af[5]=(_Float16)a1.y; af[6]=(_Float16)a1.z; af[7]=(_Float16)a1.w;
        af[8]=(_Float16)a2.x; af[9]=(_Float16)a2.y; af[10]=(_Float16)a2.z; af[11]=(_Float16)a2.w;
        af[12]=(_Float16)a3.x; af[13]=(_Float16)a3.y; af[14]=(_Float16)a3.z; af[15]=(_Float16)a3.w;
#pragma unroll
        for (int ct = 0; ct < 4; ++ct) {
            const int col  = colBase + ct * 16;
            const int colc = (col < O) ? col : (O - 1);
            const float4* wp = (const float4*)(W + (size_t)colc * Ktot + kb + (grp << 4));
            float4 w0 = wp[0], w1 = wp[1], w2 = wp[2], w3 = wp[3];
            v16h bfrag;
            bfrag[0]=(_Float16)w0.x; bfrag[1]=(_Float16)w0.y; bfrag[2]=(_Float16)w0.z; bfrag[3]=(_Float16)w0.w;
            bfrag[4]=(_Float16)w1.x; bfrag[5]=(_Float16)w1.y; bfrag[6]=(_Float16)w1.z; bfrag[7]=(_Float16)w1.w;
            bfrag[8]=(_Float16)w2.x; bfrag[9]=(_Float16)w2.y; bfrag[10]=(_Float16)w2.z; bfrag[11]=(_Float16)w2.w;
            bfrag[12]=(_Float16)w3.x; bfrag[13]=(_Float16)w3.y; bfrag[14]=(_Float16)w3.z; bfrag[15]=(_Float16)w3.w;
            acc[ct] = __builtin_amdgcn_wmma_f32_16x16x32_f16(
                false, af, false, bfrag, (short)0, acc[ct], false, false);
        }
    }

#pragma unroll
    for (int ct = 0; ct < 4; ++ct) {
        const int col = colBase + ct * 16;
        const bool cok = col < O;
        const float bsv = cok ? bias[col] : 0.f;
        float ps = 0.f, pq = 0.f;
#pragma unroll
        for (int r8 = 0; r8 < 8; ++r8) {
            int m = tm * 16 + r8 + (grp << 3);
            float v = acc[ct][r8] + bsv;
            if (cok) {
                Y[(size_t)m * ldy + col] = v;
                ps += v; pq += v * v;
            }
        }
        if (stats && cok) {
            atomicAdd(&stats[col],     ps);
            atomicAdd(&stats[O + col], pq);
        }
    }
}

// ---------------------------------------------------------------------------
// Small helpers.
// ---------------------------------------------------------------------------
__global__ void fill_kernel(float* p, int n, float v) {
    int t = blockIdx.x * blockDim.x + threadIdx.x;
    if (t < n) p[t] = v;
}

__global__ void fill_strided(float* p, int ld, int O, int rows, float v) {
    int t = blockIdx.x * blockDim.x + threadIdx.x;
    if (t >= rows * O) return;
    p[(size_t)(t / O) * ld + (t % O)] = v;
}

__global__ void bn_finalize(const float* __restrict__ stats,
                            const float* __restrict__ g,
                            const float* __restrict__ bt,
                            int O, float cnt,
                            float* __restrict__ scale, float* __restrict__ shift) {
    int o = blockIdx.x * blockDim.x + threadIdx.x;
    if (o >= O) return;
    float m   = stats[o] / cnt;
    float var = stats[O + o] / cnt - m * m;
    float s   = g[o] * rsqrtf(var + 1e-5f);
    scale[o] = s;
    shift[o] = bt[o] - m * s;
}

// Decode monotonic uint key -> float, then BN + leaky, in place (edge slices).
__global__ void bn_leaky_decode(float* __restrict__ p, int ld, int O, int rows,
                                const float* __restrict__ scale,
                                const float* __restrict__ shift) {
    int t = blockIdx.x * blockDim.x + threadIdx.x;
    if (t >= rows * O) return;
    int r = t / O, o = t % O;
    float raw = funkey(__float_as_uint(p[(size_t)r * ld + o]));
    float v = raw * scale[o] + shift[o];
    p[(size_t)r * ld + o] = leaky_f(v);
}

// Global max + mean pool over N with fused BN + leaky.
__global__ void pool_kernel(const float* __restrict__ yf,
                            const float* __restrict__ scale,
                            const float* __restrict__ shift,
                            int E, float* __restrict__ pooled) {
    int t = blockIdx.x * blockDim.x + threadIdx.x;
    if (t >= B_ * E) return;
    int b = t / E, o = t % E;
    float sc = scale[o], sh = shift[o];
    float mx = -3.4e38f, sm = 0.f;
    for (int n = 0; n < N_; ++n) {
        float v = leaky_f(yf[((size_t)b * N_ + n) * E + o] * sc + sh);
        mx = fmaxf(mx, v);
        sm += v;
    }
    pooled[(size_t)b * 2 * E + o]     = mx;
    pooled[(size_t)b * 2 * E + E + o] = sm * (1.f / N_);
}

// BN over the batch axis (16 rows) + leaky, in place. One thread per channel.
__global__ void bn_batch_leaky(float* __restrict__ Y, int R, int O,
                               const float* __restrict__ g,
                               const float* __restrict__ bt) {
    int o = blockIdx.x * blockDim.x + threadIdx.x;
    if (o >= O) return;
    float s = 0.f, sq = 0.f;
    for (int r = 0; r < R; ++r) { float v = Y[(size_t)r * O + o]; s += v; sq += v * v; }
    float m = s / R, var = sq / R - m * m;
    float sc = g[o] * rsqrtf(var + 1e-5f);
    float sh = bt[o] - m * sc;
    for (int r = 0; r < R; ++r)
        Y[(size_t)r * O + o] = leaky_f(Y[(size_t)r * O + o] * sc + sh);
}

// ---------------------------------------------------------------------------
extern "C" void kernel_launch(void* const* d_in, const int* in_sizes, int n_in,
                              void* d_out, int out_size, void* d_ws, size_t ws_size,
                              hipStream_t stream) {
    (void)in_sizes; (void)n_in; (void)out_size; (void)ws_size;

    const float* cloud = (const float*)d_in[0];
    const float *w[4], *bb[4], *gg[4], *bt[4];
    for (int i = 0; i < 4; ++i) {
        w[i]  = (const float*)d_in[1 + 4 * i];
        bb[i] = (const float*)d_in[2 + 4 * i];
        gg[i] = (const float*)d_in[3 + 4 * i];
        bt[i] = (const float*)d_in[4 + 4 * i];
    }
    const float *wf  = (const float*)d_in[17], *bfv  = (const float*)d_in[18];
    const float *gf  = (const float*)d_in[19], *btf  = (const float*)d_in[20];
    const float *wl1 = (const float*)d_in[21], *bl1  = (const float*)d_in[22];
    const float *gl1 = (const float*)d_in[23], *btl1 = (const float*)d_in[24];
    const float *wl2 = (const float*)d_in[25], *bl2  = (const float*)d_in[26];
    const float *gl2 = (const float*)d_in[27], *btl2 = (const float*)d_in[28];
    const float *wl3 = (const float*)d_in[29], *bl3  = (const float*)d_in[30];

    // Workspace carve-out (~102 MB total).
    char* base = (char*)d_ws;
    size_t off = 0;
    auto carve = [&](size_t bytes) -> void* {
        void* p = base + off;
        off += (bytes + 255) & ~(size_t)255;
        return p;
    };
    int*   nbr    = (int*)  carve((size_t)BNPTS * KNN * 4);
    float* xcat   = (float*)carve((size_t)BNPTS * LDCAT * 4);   // 33.5 MB concat
    float* yf     = (float*)carve((size_t)BNPTS * 1024 * 4);    // 67 MB final-conv out
    float* pooled = (float*)carve((size_t)B_ * 2048 * 4);
    float* fc1    = (float*)carve((size_t)B_ * 512 * 4);
    float* fc2    = (float*)carve((size_t)B_ * 256 * 4);
    float* stats  = (float*)carve(2048 * 4);
    float* scale  = (float*)carve(1024 * 4);
    float* shift  = (float*)carve(1024 * 4);

    const int MT = BNPTS * KNN / 16;   // 20480 M tiles for edge GEMMs

    // ---- Edge-conv layer 1: C=3 -> O=64 at xcat[:, 0:64] -------------------
    knn_kernel<3><<<(BNPTS + 127) / 128, 128, 0, stream>>>(cloud, 3, nbr);
    fill_strided<<<(BNPTS * 64 + 255) / 256, 256, 0, stream>>>(xcat + 0, LDCAT, 64, BNPTS, 0.f);
    fill_kernel<<<1, 128, 0, stream>>>(stats, 128, 0.f);
    edge_gemm_wmma<3, 64><<<MT, 32, 0, stream>>>(
        cloud, 3, nbr, w[0], bb[0], xcat + 0, LDCAT, stats);
    bn_finalize<<<1, 64, 0, stream>>>(stats, gg[0], bt[0], 64, (float)(BNPTS * KNN), scale, shift);
    bn_leaky_decode<<<(BNPTS * 64 + 255) / 256, 256, 0, stream>>>(xcat + 0, LDCAT, 64, BNPTS, scale, shift);

    // ---- Edge-conv layer 2: C=64 -> O=64 at xcat[:, 64:128] ----------------
    knn_kernel<64><<<(BNPTS + 127) / 128, 128, 0, stream>>>(xcat + 0, LDCAT, nbr);
    fill_strided<<<(BNPTS * 64 + 255) / 256, 256, 0, stream>>>(xcat + 64, LDCAT, 64, BNPTS, 0.f);
    fill_kernel<<<1, 128, 0, stream>>>(stats, 128, 0.f);
    edge_gemm_wmma<64, 64><<<MT, 32, 0, stream>>>(
        xcat + 0, LDCAT, nbr, w[1], bb[1], xcat + 64, LDCAT, stats);
    bn_finalize<<<1, 64, 0, stream>>>(stats, gg[1], bt[1], 64, (float)(BNPTS * KNN), scale, shift);
    bn_leaky_decode<<<(BNPTS * 64 + 255) / 256, 256, 0, stream>>>(xcat + 64, LDCAT, 64, BNPTS, scale, shift);

    // ---- Edge-conv layer 3: C=64 -> O=128 at xcat[:, 128:256] --------------
    knn_kernel<64><<<(BNPTS + 127) / 128, 128, 0, stream>>>(xcat + 64, LDCAT, nbr);
    fill_strided<<<(BNPTS * 128 + 255) / 256, 256, 0, stream>>>(xcat + 128, LDCAT, 128, BNPTS, 0.f);
    fill_kernel<<<1, 256, 0, stream>>>(stats, 256, 0.f);
    edge_gemm_wmma<64, 128><<<MT, 64, 0, stream>>>(
        xcat + 64, LDCAT, nbr, w[2], bb[2], xcat + 128, LDCAT, stats);
    bn_finalize<<<1, 128, 0, stream>>>(stats, gg[2], bt[2], 128, (float)(BNPTS * KNN), scale, shift);
    bn_leaky_decode<<<(BNPTS * 128 + 255) / 256, 256, 0, stream>>>(xcat + 128, LDCAT, 128, BNPTS, scale, shift);

    // ---- Edge-conv layer 4: C=128 -> O=256 at xcat[:, 256:512] -------------
    knn_kernel<128><<<(BNPTS + 127) / 128, 128, 0, stream>>>(xcat + 128, LDCAT, nbr);
    fill_strided<<<(BNPTS * 256 + 255) / 256, 256, 0, stream>>>(xcat + 256, LDCAT, 256, BNPTS, 0.f);
    fill_kernel<<<2, 256, 0, stream>>>(stats, 512, 0.f);
    edge_gemm_wmma<128, 256><<<MT, 128, 0, stream>>>(
        xcat + 128, LDCAT, nbr, w[3], bb[3], xcat + 256, LDCAT, stats);
    bn_finalize<<<1, 256, 0, stream>>>(stats, gg[3], bt[3], 256, (float)(BNPTS * KNN), scale, shift);
    bn_leaky_decode<<<(BNPTS * 256 + 255) / 256, 256, 0, stream>>>(xcat + 256, LDCAT, 256, BNPTS, scale, shift);

    // ---- Final 1x1 conv 512 -> 1024, BN stats, fused pool ------------------
    fill_kernel<<<8, 256, 0, stream>>>(stats, 2048, 0.f);
    {
        int waves = (BNPTS / 16) * (1024 / 64);   // 16384 waves
        gemm_wmma<512><<<(waves + 3) / 4, 128, 0, stream>>>(
            xcat, LDCAT, BNPTS, wf, bfv, 1024, yf, 1024, stats);
    }
    bn_finalize<<<4, 256, 0, stream>>>(stats, gf, btf, 1024, (float)BNPTS, scale, shift);
    pool_kernel<<<(B_ * 1024 + 255) / 256, 256, 0, stream>>>(yf, scale, shift, 1024, pooled);

    // ---- FC head: 2048 -> 512 -> 256 -> 40 ---------------------------------
    gemm_wmma<2048><<<2, 128, 0, stream>>>(pooled, 2048, B_, wl1, bl1, 512, fc1, 512, nullptr);
    bn_batch_leaky<<<2, 256, 0, stream>>>(fc1, B_, 512, gl1, btl1);
    gemm_wmma<512><<<1, 128, 0, stream>>>(fc1, 512, B_, wl2, bl2, 256, fc2, 256, nullptr);
    bn_batch_leaky<<<1, 256, 0, stream>>>(fc2, B_, 256, gl2, btl2);
    gemm_wmma<256><<<1, 128, 0, stream>>>(fc2, 256, B_, wl3, bl3, 40, (float*)d_out, 40, nullptr);
}
